// WindowAttention_90108413870631
// MI455X (gfx1250) — compile-verified, matrix operand order checked by reading, not run on hardware
//
#include <hip/hip_runtime.h>
#include <hip/hip_bf16.h>

// ---------------------------------------------------------------------------
// Window attention (Swin-style) for MI455X / gfx1250, wave32 + WMMA f16->f32.
//   DIM=192 HEADS=6 hd=32 Nq=Nkv=192 B_=512
// Roofline: ~43 GFLOP vs ~300MB HBM traffic -> balanced at f16-WMMA rates;
// all GEMMs use v_wmma_f32_16x16x32_f16 (K=32 == head_dim).
// This revision prefetches all 12 B-fragments per k-step into registers
// (load clause) before the 12 independent WMMAs, so loads pipeline instead of
// the load->wait(0)->wmma lockstep seen in the previous disassembly.
// ---------------------------------------------------------------------------

#define DIMC   192
#define HEADS  6
#define HD     32
#define NWIN   64
#define NQ     192
#define BWIN   512
#define QSCALE 0.17677669529663687f   // 32^-0.5

typedef __attribute__((ext_vector_type(16))) _Float16 v16h;
typedef __attribute__((ext_vector_type(8)))  float    v8f;
typedef __attribute__((ext_vector_type(8)))  _Float16 v8h;

__device__ __forceinline__ v8f vzero8() {
  v8f z;
#pragma unroll
  for (int i = 0; i < 8; ++i) z[i] = 0.0f;
  return z;
}

// Load a 16x32 f16 WMMA A/B fragment from a row-major tile (Bt row-major for B).
// Lane l: row = l&15, K-chunks [half*8 .. +7] and [16+half*8 .. +7] (ISA 7.12.2).
__device__ __forceinline__ v16h load_frag(const _Float16* tile, int ld) {
  const int lane = threadIdx.x & 31;
  const _Float16* p = tile + (lane & 15) * ld + (lane >> 4) * 8;
  v8h lo = *(const v8h*)(p);
  v8h hi = *(const v8h*)(p + 16);
  v16h r;
#pragma unroll
  for (int i = 0; i < 8; ++i) { r[i] = lo[i]; r[i + 8] = hi[i]; }
  return r;
}

__device__ __forceinline__ v8f wmma16(v16h a, v16h b, v8f c) {
  // (neg_a, A, neg_b, B, c_mod, C, reuse_a, reuse_b)
  return __builtin_amdgcn_wmma_f32_16x16x32_f16(false, a, false, b, (short)0, c,
                                                false, false);
}

// ---------------------------------------------------------------------------
// Kernel 1: weights f32->f16 (layout already B^T-friendly) + fused rel-pos bias
// table gather biasM[h][q64][k64] = table[rel[q64*64+k64]*HEADS + h].
// ---------------------------------------------------------------------------
__global__ void prep_kernel(const float* __restrict__ Wq, const float* __restrict__ Wk,
                            const float* __restrict__ Wv, const float* __restrict__ Wp,
                            const float* __restrict__ table, const int* __restrict__ rel,
                            _Float16* __restrict__ Whq, _Float16* __restrict__ Whk,
                            _Float16* __restrict__ Whv, _Float16* __restrict__ Whp,
                            float* __restrict__ biasM) {
  const int i = blockIdx.x * blockDim.x + threadIdx.x;
  if (i < DIMC * DIMC) {
    Whq[i] = (_Float16)Wq[i];
    Whk[i] = (_Float16)Wk[i];
    Whv[i] = (_Float16)Wv[i];
    Whp[i] = (_Float16)Wp[i];
  }
  if (i < HEADS * NWIN * NWIN) {
    const int h  = i / (NWIN * NWIN);
    const int qk = i % (NWIN * NWIN);
    biasM[i] = table[rel[qk] * HEADS + h];
  }
}

// ---------------------------------------------------------------------------
// Kernel 2: q/k/v projections. grid=(768,1,3) block=256 (8 waves).
// Each block: 128 rows x 192 cols; wave w owns rows [w*16, w*16+16).
// A fragments read directly from f32 global (each x element read once) and
// converted to f16; B fragments prefetched (12 frags = 24 b128 loads in a
// clause) from pre-converted f16 weights (L2-hot), then 12 independent WMMAs.
// Outputs (f16): q,k -> (b,h,n,d) [B^T layout for Q@K^T], v -> (b,h,d,n).
// ---------------------------------------------------------------------------
__global__ __launch_bounds__(256) void qkv_proj_kernel(
    const float* __restrict__ xq, const float* __restrict__ xk, const float* __restrict__ xv,
    const _Float16* __restrict__ Whq, const _Float16* __restrict__ Whk,
    const _Float16* __restrict__ Whv,
    const float* __restrict__ bq, const float* __restrict__ bk, const float* __restrict__ bv,
    _Float16* __restrict__ qo, _Float16* __restrict__ ko, _Float16* __restrict__ vo) {
  const int mode = blockIdx.z;  // 0=q 1=k 2=v
  const float*    x    = (mode == 0) ? xq : (mode == 1) ? xk : xv;
  const _Float16* W    = (mode == 0) ? Whq : (mode == 1) ? Whk : Whv;
  const float*    bias = (mode == 0) ? bq : (mode == 1) ? bk : bv;

  const int lane  = threadIdx.x & 31;
  const int wave  = threadIdx.x >> 5;
  const int hf    = lane >> 4;
  const int n16   = lane & 15;
  const int mrow0 = blockIdx.x * 128 + wave * 16;
  const int arow  = mrow0 + n16;

  v8f acc[12];
#pragma unroll
  for (int t = 0; t < 12; ++t) acc[t] = vzero8();

  for (int ks = 0; ks < 6; ++ks) {
    const float* ap = x + (size_t)arow * DIMC + ks * 32 + hf * 8;
    float4 f0 = *(const float4*)(ap);
    float4 f1 = *(const float4*)(ap + 4);
    float4 f2 = *(const float4*)(ap + 16);
    float4 f3 = *(const float4*)(ap + 20);
    v16h a;
    a[0]  = (_Float16)f0.x; a[1]  = (_Float16)f0.y; a[2]  = (_Float16)f0.z; a[3]  = (_Float16)f0.w;
    a[4]  = (_Float16)f1.x; a[5]  = (_Float16)f1.y; a[6]  = (_Float16)f1.z; a[7]  = (_Float16)f1.w;
    a[8]  = (_Float16)f2.x; a[9]  = (_Float16)f2.y; a[10] = (_Float16)f2.z; a[11] = (_Float16)f2.w;
    a[12] = (_Float16)f3.x; a[13] = (_Float16)f3.y; a[14] = (_Float16)f3.z; a[15] = (_Float16)f3.w;

    v16h bfr[12];
#pragma unroll
    for (int t = 0; t < 12; ++t)
      bfr[t] = load_frag(W + t * 16 * DIMC + ks * 32, DIMC);
#pragma unroll
    for (int t = 0; t < 12; ++t)
      acc[t] = wmma16(a, bfr[t], acc[t]);
  }

#pragma unroll
  for (int r = 0; r < 8; ++r) {
    const int m  = mrow0 + r + 8 * hf;   // global row in (B_*Nq)
    const int bb = m / NQ;
    const int n  = m % NQ;
#pragma unroll
    for (int t = 0; t < 12; ++t) {
      const int h = t >> 1;
      const int d = (t & 1) * 16 + n16;
      const int c = h * HD + d;
      const float val = acc[t][r] + bias[c];
      if (mode == 0) {
        qo[(((size_t)bb * HEADS + h) * NQ + n) * HD + d] = (_Float16)(val * QSCALE);
      } else if (mode == 1) {
        ko[(((size_t)bb * HEADS + h) * NQ + n) * HD + d] = (_Float16)val;
      } else {
        vo[(((size_t)bb * HEADS + h) * HD + d) * NQ + n] = (_Float16)val;  // transposed
      }
    }
  }
}

// ---------------------------------------------------------------------------
// Kernel 3: fused attention. grid=3072 (one per (b,h)), block=128 (4 waves).
// K (192x32) and V^T (32x192) staged in padded LDS; each wave handles 3 query
// tiles: S row (12 tiles) via 1 WMMA/tile (K=32==hd) with all 12 LDS
// B-fragments prefetched first, then fused bias+mask+softmax (shfl_xor
// reductions), P->LDS (f16), O = P@V via WMMA.
// ---------------------------------------------------------------------------
__global__ __launch_bounds__(128) void attn_kernel(
    const _Float16* __restrict__ q, const _Float16* __restrict__ k,
    const _Float16* __restrict__ vT, const float* __restrict__ biasM,
    const float* __restrict__ mask, _Float16* __restrict__ O) {
  const int bh = blockIdx.x;
  const int b  = bh / HEADS;
  const int h  = bh % HEADS;

  __shared__ _Float16 k_s[NQ * 40];        // 192 kv rows, ld=40 (pad: conflict-free b128)
  __shared__ _Float16 v_s[HD * 200];       // 32 d rows,  ld=200
  __shared__ _Float16 p_s[4 * 16 * 200];   // per-wave P tile (16 x 192), ld=200

  const int tid  = threadIdx.x;
  const int lane = tid & 31;
  const int wave = tid >> 5;
  const int hf   = lane >> 4;
  const int n16  = lane & 15;

  const _Float16* kg = k  + (size_t)bh * NQ * HD;
  const _Float16* vg = vT + (size_t)bh * HD * NQ;
  const _Float16* qg = q  + (size_t)bh * NQ * HD;

  for (int r = tid; r < NQ; r += 128) {
    const v8h* src = (const v8h*)(kg + r * HD);
    v8h* dst = (v8h*)(k_s + r * 40);
    dst[0] = src[0]; dst[1] = src[1]; dst[2] = src[2]; dst[3] = src[3];
  }
  for (int i = tid; i < HD * 24; i += 128) {
    const int r = i / 24, ch = i % 24;
    *(v8h*)(v_s + r * 200 + ch * 8) = *(const v8h*)(vg + r * NQ + ch * 8);
  }
  __syncthreads();

  _Float16* pw = p_s + wave * 16 * 200;

  for (int qi = 0; qi < 3; ++qi) {
    const int qt = wave * 3 + qi;
    const v16h aq = load_frag(qg + qt * 16 * HD, HD);

    v8f acc[12];
#pragma unroll
    for (int t = 0; t < 12; ++t) acc[t] = vzero8();
    {
      v16h bfr[12];
#pragma unroll
      for (int t = 0; t < 12; ++t)
        bfr[t] = load_frag(k_s + t * 16 * 40, 40);
#pragma unroll
      for (int t = 0; t < 12; ++t)
        acc[t] = wmma16(aq, bfr[t], acc[t]);
    }

    // fused rel-pos bias + window mask + row softmax
#pragma unroll
    for (int r = 0; r < 8; ++r) {
      const int qrow = qt * 16 + r + 8 * hf;
      const int q64  = qrow & 63;
      const float* bpp = biasM + ((size_t)h * NWIN + q64) * NWIN;
      const float* mpp = mask  + ((size_t)b * NWIN + q64) * NWIN;
      float sv[12];
      float mx = -3.0e30f;
#pragma unroll
      for (int t = 0; t < 12; ++t) {
        const int k64 = (t * 16 + n16) & 63;
        const float s = acc[t][r] + bpp[k64] + mpp[k64];
        sv[t] = s;
        mx = fmaxf(mx, s);
      }
      mx = fmaxf(mx, __shfl_xor(mx, 1));
      mx = fmaxf(mx, __shfl_xor(mx, 2));
      mx = fmaxf(mx, __shfl_xor(mx, 4));
      mx = fmaxf(mx, __shfl_xor(mx, 8));
      float sum = 0.0f;
#pragma unroll
      for (int t = 0; t < 12; ++t) {
        const float e = __expf(sv[t] - mx);
        sv[t] = e;
        sum += e;
      }
      sum += __shfl_xor(sum, 1);
      sum += __shfl_xor(sum, 2);
      sum += __shfl_xor(sum, 4);
      sum += __shfl_xor(sum, 8);
      const float inv = 1.0f / sum;
      const int prow = r + 8 * hf;
#pragma unroll
      for (int t = 0; t < 12; ++t)
        pw[prow * 200 + t * 16 + n16] = (_Float16)(sv[t] * inv);
    }
    __syncthreads();  // uniform (3 iters/wave); orders P stores vs frag loads

    v8f o0 = vzero8(), o1 = vzero8();
#pragma unroll
    for (int ks = 0; ks < 6; ++ks) {
      v16h apf = load_frag(pw + ks * 32, 200);
      v16h b0  = load_frag(v_s + ks * 32, 200);
      v16h b1  = load_frag(v_s + 16 * 200 + ks * 32, 200);
      o0 = wmma16(apf, b0, o0);
      o1 = wmma16(apf, b1, o1);
    }
#pragma unroll
    for (int r = 0; r < 8; ++r) {
      const int qrow = qt * 16 + r + 8 * hf;
      _Float16* op = O + ((size_t)b * NQ + qrow) * DIMC + h * HD;
      op[n16]      = (_Float16)o0[r];
      op[16 + n16] = (_Float16)o1[r];
    }
  }
}

// ---------------------------------------------------------------------------
// Kernel 4: output projection, f16 A direct from global, B-fragments
// prefetched per k-step, f32 result to d_out
// (layout (b, n, c) == (B_, NC, 8, 8, C) contiguous).
// ---------------------------------------------------------------------------
__global__ __launch_bounds__(256) void out_proj_kernel(
    const _Float16* __restrict__ O, const _Float16* __restrict__ Whp,
    const float* __restrict__ bp, float* __restrict__ out) {
  const int lane  = threadIdx.x & 31;
  const int wave  = threadIdx.x >> 5;
  const int hf    = lane >> 4;
  const int n16   = lane & 15;
  const int mrow0 = blockIdx.x * 128 + wave * 16;

  v8f acc[12];
#pragma unroll
  for (int t = 0; t < 12; ++t) acc[t] = vzero8();

  for (int ks = 0; ks < 6; ++ks) {
    v16h a = load_frag(O + (size_t)mrow0 * DIMC + ks * 32, DIMC);
    v16h bfr[12];
#pragma unroll
    for (int t = 0; t < 12; ++t)
      bfr[t] = load_frag(Whp + t * 16 * DIMC + ks * 32, DIMC);
#pragma unroll
    for (int t = 0; t < 12; ++t)
      acc[t] = wmma16(a, bfr[t], acc[t]);
  }
#pragma unroll
  for (int r = 0; r < 8; ++r) {
    const int m = mrow0 + r + 8 * hf;
#pragma unroll
    for (int t = 0; t < 12; ++t) {
      const int c = t * 16 + n16;
      out[(size_t)m * DIMC + c] = acc[t][r] + bp[c];
    }
  }
}

// ---------------------------------------------------------------------------
extern "C" void kernel_launch(void* const* d_in, const int* in_sizes, int n_in,
                              void* d_out, int out_size, void* d_ws, size_t ws_size,
                              hipStream_t stream) {
  (void)in_sizes; (void)n_in; (void)out_size; (void)ws_size;
  const float* xq    = (const float*)d_in[0];
  const float* xk    = (const float*)d_in[1];
  const float* xv    = (const float*)d_in[2];
  const float* mask  = (const float*)d_in[3];
  const float* Wq    = (const float*)d_in[4];
  const float* bq    = (const float*)d_in[5];
  const float* Wk    = (const float*)d_in[6];
  const float* bk    = (const float*)d_in[7];
  const float* Wv    = (const float*)d_in[8];
  const float* bv    = (const float*)d_in[9];
  const float* Wp    = (const float*)d_in[10];
  const float* bp    = (const float*)d_in[11];
  const float* table = (const float*)d_in[12];
  const int*   rel   = (const int*)d_in[13];
  float* out = (float*)d_out;

  char* ws = (char*)d_ws;
  auto take = [&](size_t bytes) -> char* {
    char* p = ws;
    ws += (bytes + 255) & ~(size_t)255;
    return p;
  };
  const size_t qkvElems = (size_t)BWIN * HEADS * NQ * HD;  // 18,874,368
  _Float16* Whq   = (_Float16*)take((size_t)DIMC * DIMC * 2);
  _Float16* Whk   = (_Float16*)take((size_t)DIMC * DIMC * 2);
  _Float16* Whv   = (_Float16*)take((size_t)DIMC * DIMC * 2);
  _Float16* Whp   = (_Float16*)take((size_t)DIMC * DIMC * 2);
  float*    biasM = (float*)take((size_t)HEADS * NWIN * NWIN * 4);
  _Float16* qbuf  = (_Float16*)take(qkvElems * 2);
  _Float16* kbuf  = (_Float16*)take(qkvElems * 2);
  _Float16* vbuf  = (_Float16*)take(qkvElems * 2);   // transposed (b,h,d,n)
  _Float16* obuf  = (_Float16*)take((size_t)BWIN * NQ * DIMC * 2);

  prep_kernel<<<144, 256, 0, stream>>>(Wq, Wk, Wv, Wp, table, rel,
                                       Whq, Whk, Whv, Whp, biasM);

  dim3 gproj((BWIN * NQ) / 128, 1, 3);  // 768 x 1 x 3
  qkv_proj_kernel<<<gproj, 256, 0, stream>>>(xq, xk, xv, Whq, Whk, Whv,
                                             bq, bk, bv, qbuf, kbuf, vbuf);

  attn_kernel<<<BWIN * HEADS, 128, 0, stream>>>(qbuf, kbuf, vbuf, biasM, mask, obuf);

  out_proj_kernel<<<(BWIN * NQ) / 128, 256, 0, stream>>>(obuf, Whp, bp, out);
}
